// CrossMambaFusion_73976516706904
// MI455X (gfx1250) — compile-verified, matrix operand order checked by reading, not run on hardware
//
#include <hip/hip_runtime.h>
#include <hip/hip_bf16.h>
#include <math.h>

typedef __bf16 bf16_t;
typedef __attribute__((ext_vector_type(16))) __bf16 v16bf;
typedef __attribute__((ext_vector_type(8)))  __bf16 bfx8;
typedef __attribute__((ext_vector_type(8)))  float  v8f;

static constexpr int BATCH = 4;
static constexpr int LSEQ  = 4096;           // H*W
static constexpr int TSEQ  = 2 * LSEQ;       // 8192 interleaved (x, skip)
static constexpr int CDIM  = 256;            // D_MODEL
static constexpr int DIN   = 512;            // D_INNER
static constexpr int NST   = 16;             // D_STATE
static constexpr int BT    = BATCH * TSEQ;   // 32768 rows
static constexpr int BL    = BATCH * LSEQ;   // 16384 even rows
static constexpr int CHT   = 128;            // scan chunk length
static constexpr int NCH   = TSEQ / CHT;     // 64 chunks per batch

// ============== CDNA5 async global->LDS staging ======================
__device__ __forceinline__ void async_b128_to_lds(const void* gptr, void* lptr)
{
    unsigned long long g = (unsigned long long)(uintptr_t)gptr;
    unsigned int       l = (unsigned int)(uintptr_t)lptr;   // LDS byte offset
    asm volatile("global_load_async_to_lds_b128 %0, %1, off"
                 :: "v"(l), "v"(g) : "memory");
}
__device__ __forceinline__ void wait_async0()
{
    asm volatile("s_wait_asynccnt 0x0" ::: "memory");
}

// ================= WMMA helpers (bf16 in, f32 accum) =================
// 16-bit A-frag layout (ISA 7.12.2): lanes 0-15 -> M=0..15, K=0..7 & 16..23
//                                    lanes 16-31 -> same M,  K=8..15 & 24..31
__device__ __forceinline__ v16bf load_frag(const bf16_t* mat, int row0, int ld,
                                           int k0, int lane)
{
    const int m  = lane & 15;
    const int hi = lane >> 4;
    const bf16_t* p = mat + (size_t)(row0 + m) * ld + k0 + hi * 8;
    bfx8 lo = *(const bfx8*)(p);        // K = hi*8 .. hi*8+7
    bfx8 hh = *(const bfx8*)(p + 16);   // K = 16+hi*8 .. 16+hi*8+7
    v16bf r;
#pragma unroll
    for (int i = 0; i < 8; ++i) { r[i] = lo[i]; r[i + 8] = hh[i]; }
    return r;
}

// 16*MT x 16*NACC wave tile, register double-buffered over K.
template <int MT, int NACC, int K>
__device__ __forceinline__ void gemm_tile(const bf16_t* A, const bf16_t* Bw,
                                          int m0, int n0, int lane,
                                          v8f acc[MT][NACC])
{
#pragma unroll
    for (int mi = 0; mi < MT; ++mi)
#pragma unroll
        for (int j = 0; j < NACC; ++j)
#pragma unroll
            for (int i = 0; i < 8; ++i) acc[mi][j][i] = 0.f;

    v16bf af[MT], bfr[NACC];
#pragma unroll
    for (int mi = 0; mi < MT; ++mi) af[mi] = load_frag(A, m0 + 16 * mi, K, 0, lane);
#pragma unroll
    for (int j = 0; j < NACC; ++j)  bfr[j] = load_frag(Bw, n0 + 16 * j, K, 0, lane);

#pragma unroll
    for (int k0 = 0; k0 < K; k0 += 32) {
        v16bf af2[MT], bf2[NACC];
        if (k0 + 32 < K) {            // prefetch next K-tile into fresh regs
#pragma unroll
            for (int mi = 0; mi < MT; ++mi)
                af2[mi] = load_frag(A, m0 + 16 * mi, K, k0 + 32, lane);
#pragma unroll
            for (int j = 0; j < NACC; ++j)
                bf2[j] = load_frag(Bw, n0 + 16 * j, K, k0 + 32, lane);
        }
#pragma unroll
        for (int mi = 0; mi < MT; ++mi)
#pragma unroll
            for (int j = 0; j < NACC; ++j)
                acc[mi][j] = __builtin_amdgcn_wmma_f32_16x16x32_bf16(
                    false, af[mi], false, bfr[j], (short)0, acc[mi][j],
                    false, false);
        if (k0 + 32 < K) {
#pragma unroll
            for (int mi = 0; mi < MT; ++mi) af[mi] = af2[mi];
#pragma unroll
            for (int j = 0; j < NACC; ++j)  bfr[j] = bf2[j];
        }
    }
}

template <int MT>
__device__ __forceinline__ void tile_coords(int tilesN, int NT,
                                            int& m0, int& n0, int& lane)
{
    const int wid = blockIdx.x * (blockDim.x >> 5) + (threadIdx.x >> 5);
    lane = threadIdx.x & 31;
    m0 = (wid / tilesN) * (16 * MT);
    n0 = (wid % tilesN) * NT;
}

// ===================== GEMM kernels ==================================
// xz[t,e] = sum_c inter[t,c] * Win[e,c];  M=BT, N=1024, K=256
__global__ void gemm_inproj(const bf16_t* __restrict__ A,
                            const bf16_t* __restrict__ Bw,
                            float* __restrict__ Cm)
{
    int m0, n0, lane; tile_coords<2>(1024 / 64, 64, m0, n0, lane);
    v8f acc[2][4];
    gemm_tile<2, 4, CDIM>(A, Bw, m0, n0, lane, acc);
    const int hi = lane >> 4, nn = lane & 15;
#pragma unroll
    for (int mi = 0; mi < 2; ++mi)
#pragma unroll
        for (int j = 0; j < 4; ++j)
#pragma unroll
            for (int i = 0; i < 8; ++i)
                Cm[(size_t)(m0 + 16 * mi + i + hi * 8) * 1024 + n0 + 16 * j + nn]
                    = acc[mi][j][i];
}

// x_dbl[t,e] = sum_d u[t,d] * Wx[e,d];  M=BT, N=48 (3x16 tiles), K=512
__global__ void gemm_xproj(const bf16_t* __restrict__ A,
                           const bf16_t* __restrict__ Bw,
                           float* __restrict__ Cm)
{
    int m0, n0, lane; tile_coords<2>(1, 48, m0, n0, lane);
    v8f acc[2][3];
    gemm_tile<2, 3, DIN>(A, Bw, m0, n0, lane, acc);
    const int hi = lane >> 4, nn = lane & 15;
#pragma unroll
    for (int mi = 0; mi < 2; ++mi)
#pragma unroll
        for (int j = 0; j < 3; ++j)
#pragma unroll
            for (int i = 0; i < 8; ++i)
                Cm[(size_t)(m0 + 16 * mi + i + hi * 8) * 48 + 16 * j + nn]
                    = acc[mi][j][i];
}

// dt[t,d] = softplus(sum_r dtr[t,r]*Wdt[d,r] + b[d]);  M=BT, N=512, K=32(pad)
__global__ void gemm_dtproj(const bf16_t* __restrict__ A,
                            const bf16_t* __restrict__ Bw,
                            const float* __restrict__ bias,
                            float* __restrict__ Cm)
{
    int m0, n0, lane; tile_coords<2>(512 / 64, 64, m0, n0, lane);
    v8f acc[2][4];
    gemm_tile<2, 4, 32>(A, Bw, m0, n0, lane, acc);
    const int hi = lane >> 4, nn = lane & 15;
#pragma unroll
    for (int mi = 0; mi < 2; ++mi)
#pragma unroll
        for (int j = 0; j < 4; ++j) {
            const int col = n0 + 16 * j + nn;
            const float bv = bias[col];
#pragma unroll
            for (int i = 0; i < 8; ++i) {
                float v = acc[mi][j][i] + bv;
                v = (v > 20.f) ? v : log1pf(__expf(v));   // softplus
                Cm[(size_t)(m0 + 16 * mi + i + hi * 8) * 512 + col] = v;
            }
        }
}

// mo[l,c] = sum_d y_even[l,d]*Wm[c,d];  M=BL, N=256, K=512 -> bf16 out
__global__ void gemm_mo(const bf16_t* __restrict__ A,
                        const bf16_t* __restrict__ Bw,
                        bf16_t* __restrict__ Cm)
{
    int m0, n0, lane; tile_coords<2>(256 / 64, 64, m0, n0, lane);
    v8f acc[2][4];
    gemm_tile<2, 4, DIN>(A, Bw, m0, n0, lane, acc);
    const int hi = lane >> 4, nn = lane & 15;
#pragma unroll
    for (int mi = 0; mi < 2; ++mi)
#pragma unroll
        for (int j = 0; j < 4; ++j)
#pragma unroll
            for (int i = 0; i < 8; ++i)
                Cm[(size_t)(m0 + 16 * mi + i + hi * 8) * 256 + n0 + 16 * j + nn]
                    = (bf16_t)acc[mi][j][i];
}

// out[l,d] = sum_c mo[l,c]*Wo[d,c] + ob[d] + LN(x)[l,d];  M=BL, N=256, K=256
__global__ void gemm_out(const bf16_t* __restrict__ A,
                         const bf16_t* __restrict__ Bw,
                         const float* __restrict__ ob,
                         const float* __restrict__ inter_f,
                         float* __restrict__ out)
{
    int m0, n0, lane; tile_coords<2>(256 / 64, 64, m0, n0, lane);
    v8f acc[2][4];
    gemm_tile<2, 4, CDIM>(A, Bw, m0, n0, lane, acc);
    const int hi = lane >> 4, nn = lane & 15;
#pragma unroll
    for (int mi = 0; mi < 2; ++mi)
#pragma unroll
        for (int j = 0; j < 4; ++j) {
            const int col = n0 + 16 * j + nn;
            const float bv = ob[col];
#pragma unroll
            for (int i = 0; i < 8; ++i) {
                const size_t row = (size_t)(m0 + 16 * mi + i + hi * 8);
                // residual: x_flat row l lives at interleaved row 2*l
                out[row * 256 + col] =
                    acc[mi][j][i] + bv + inter_f[(row * 2) * 256 + col];
            }
        }
}

// ===================== elementwise / LN / conv =======================
__device__ __forceinline__ void ln_row(const float* __restrict__ src,
                                       const float* __restrict__ w,
                                       const float* __restrict__ b,
                                       float* __restrict__ dstf,
                                       bf16_t* __restrict__ dstb, int lane)
{
    float v[8], s = 0.f, ss = 0.f;
#pragma unroll
    for (int i = 0; i < 8; ++i) {
        float t = src[lane + 32 * i];
        v[i] = t; s += t; ss += t * t;
    }
#pragma unroll
    for (int o = 16; o > 0; o >>= 1) {
        s  += __shfl_xor(s,  o, 32);
        ss += __shfl_xor(ss, o, 32);
    }
    const float mu  = s * (1.f / 256.f);
    const float var = ss * (1.f / 256.f) - mu * mu;
    const float r   = rsqrtf(var + 1e-5f);
#pragma unroll
    for (int i = 0; i < 8; ++i) {
        const int c = lane + 32 * i;
        const float o = (v[i] - mu) * r * w[c] + b[c];
        dstf[c] = o; dstb[c] = (bf16_t)o;
    }
}

// One wave per spatial position: LN(x) -> inter row 2p, LN(skip) -> 2p+1
__global__ void ln_interleave(const float* __restrict__ x,
                              const float* __restrict__ skip,
                              const float* __restrict__ wx, const float* __restrict__ bx,
                              const float* __restrict__ ws, const float* __restrict__ bs,
                              float* __restrict__ inter_f, bf16_t* __restrict__ inter_b)
{
    const int lane = threadIdx.x & 31;
    const int p = blockIdx.x * 8 + (threadIdx.x >> 5);   // 0..BL-1
    ln_row(x    + (size_t)p * 256, wx, bx,
           inter_f + (size_t)(2 * p) * 256, inter_b + (size_t)(2 * p) * 256, lane);
    ln_row(skip + (size_t)p * 256, ws, bs,
           inter_f + (size_t)(2 * p + 1) * 256, inter_b + (size_t)(2 * p + 1) * 256, lane);
}

// causal depthwise conv (k=4) + bias + silu; u half of xz
__global__ void conv_silu(const float* __restrict__ xz,
                          const float* __restrict__ cw,
                          const float* __restrict__ cb,
                          float* __restrict__ us, bf16_t* __restrict__ ub)
{
    const size_t idx = (size_t)blockIdx.x * 256 + threadIdx.x;  // BT*512
    const int d = (int)(idx & 511);
    const size_t bt = idx >> 9;
    const int t = (int)(bt & (TSEQ - 1));
    float acc = cb[d];
#pragma unroll
    for (int k = 0; k < 4; ++k) {
        const int tt = t - 3 + k;
        if (tt >= 0)
            acc = fmaf(cw[d * 4 + k], xz[(bt - 3 + k) * 1024 + d], acc);
    }
    const float s = acc / (1.f + __expf(-acc));
    us[idx] = s;
    ub[idx] = (bf16_t)s;
}

// extract dt-rank columns of x_dbl into K=32 zero-padded bf16 A-matrix
__global__ void dtprep(const float* __restrict__ xdbl, bf16_t* __restrict__ dtb)
{
    const int idx = blockIdx.x * 256 + threadIdx.x;   // BT*32
    const int r = idx & 31;
    const size_t bt = (size_t)(idx >> 5);
    dtb[idx] = (bf16_t)((r < 16) ? xdbl[bt * 48 + r] : 0.f);
}

__global__ void f32_to_bf16(const float* __restrict__ s, bf16_t* __restrict__ d, int n)
{
    const int i = blockIdx.x * 256 + threadIdx.x;
    if (i < n) d[i] = (bf16_t)s[i];
}

__global__ void pad_dtw(const float* __restrict__ s, bf16_t* __restrict__ d)
{
    const int i = blockIdx.x * 256 + threadIdx.x;     // 512*32
    const int k = i & 31, row = i >> 5;
    d[i] = (bf16_t)((k < 16) ? s[row * 16 + k] : 0.f);
}

// ===================== chunked parallel scan =========================
// Pass A: per-chunk A-product and zero-state solution.
__global__ void scan_partial(const float* __restrict__ dts,
                             const float* __restrict__ us,
                             const float* __restrict__ xdbl,
                             const float* __restrict__ A_log,
                             float* __restrict__ chA, float* __restrict__ chB)
{
    __shared__ float Bsh[CHT * NST];
    const int b = blockIdx.x >> 6;          // NCH = 64
    const int g = blockIdx.x & 63;
    const int d = threadIdx.x;              // 512
    const int t0 = g * CHT;
    {   // async-stage the chunk's B_t values: one b128 per thread
        const int tt = threadIdx.x >> 2;            // 0..127
        const int part = (threadIdx.x & 3) * 4;     // 0,4,8,12
        async_b128_to_lds(&xdbl[((size_t)b * TSEQ + t0 + tt) * 48 + 16 + part],
                          &Bsh[tt * NST + part]);
        wait_async0();
    }
    __syncthreads();

    float Ar[NST], pA[NST], pB[NST];
#pragma unroll
    for (int n = 0; n < NST; ++n) {
        Ar[n] = -__expf(A_log[d * NST + n]);
        pA[n] = 1.f; pB[n] = 0.f;
    }
#pragma unroll 2
    for (int tt = 0; tt < CHT; ++tt) {
        const size_t bt = (size_t)b * TSEQ + t0 + tt;
        const float dt = dts[bt * 512 + d];
        const float du = dt * us[bt * 512 + d];
#pragma unroll
        for (int n = 0; n < NST; ++n) {
            const float a = __expf(dt * Ar[n]);
            pB[n] = fmaf(a, pB[n], du * Bsh[tt * NST + n]);
            pA[n] *= a;
        }
    }
    const size_t base = (((size_t)b * NCH + g) * DIN + d) * NST;
#pragma unroll
    for (int n = 0; n < NST; ++n) { chA[base + n] = pA[n]; chB[base + n] = pB[n]; }
}

// Pass B: 64-step prefix over chunk boundaries (one lane per (b,d,n)).
// Addresses are recurrence-independent -> unroll so loads pipeline ahead
// of the dependent FMA chain.
__global__ void scan_prefix(const float* __restrict__ chA,
                            const float* __restrict__ chB,
                            float* __restrict__ hst)
{
    const int tid = blockIdx.x * 256 + threadIdx.x;   // BATCH*DIN*NST = 32768
    const int b = tid >> 13;
    const int dn = tid & 8191;
    float h = 0.f;
#pragma unroll 8
    for (int g = 0; g < NCH; ++g) {
        const size_t idx = (((size_t)b * NCH + g) * DIN * NST) + dn;
        hst[idx] = h;
        h = chA[idx] * h + chB[idx];
    }
}

// Pass C: replay chunks with correct h0; fuse +u*D, *silu(z); keep even t.
__global__ void scan_final(const float* __restrict__ dts,
                           const float* __restrict__ us,
                           const float* __restrict__ xdbl,
                           const float* __restrict__ A_log,
                           const float* __restrict__ Dv,
                           const float* __restrict__ xz,
                           const float* __restrict__ hst,
                           bf16_t* __restrict__ yev)
{
    __shared__ float Bsh[CHT * NST];
    __shared__ float Csh[CHT * NST];
    const int b = blockIdx.x >> 6;
    const int g = blockIdx.x & 63;
    const int d = threadIdx.x;
    const int t0 = g * CHT;
    {   // async-stage B_t and C_t for the whole chunk
        const int tt = threadIdx.x >> 2;
        const int part = (threadIdx.x & 3) * 4;
        const size_t r = ((size_t)b * TSEQ + t0 + tt) * 48;
        async_b128_to_lds(&xdbl[r + 16 + part], &Bsh[tt * NST + part]);
        async_b128_to_lds(&xdbl[r + 32 + part], &Csh[tt * NST + part]);
        wait_async0();
    }
    __syncthreads();

    float Ar[NST], h[NST];
    const size_t hbase = (((size_t)b * NCH + g) * DIN + d) * NST;
#pragma unroll
    for (int n = 0; n < NST; ++n) {
        Ar[n] = -__expf(A_log[d * NST + n]);
        h[n]  = hst[hbase + n];
    }
    const float Dd = Dv[d];
#pragma unroll 2
    for (int tt = 0; tt < CHT; ++tt) {
        const size_t bt = (size_t)b * TSEQ + t0 + tt;
        const float dt = dts[bt * 512 + d];
        const float u  = us[bt * 512 + d];
        const float du = dt * u;
        float y = 0.f;
#pragma unroll
        for (int n = 0; n < NST; ++n) {
            const float a = __expf(dt * Ar[n]);
            h[n] = fmaf(a, h[n], du * Bsh[tt * NST + n]);
            y = fmaf(h[n], Csh[tt * NST + n], y);
        }
        const int t = t0 + tt;
        if ((t & 1) == 0) {                       // only even t survives
            const float z = xz[bt * 1024 + 512 + d];
            const float sil = z / (1.f + __expf(-z));
            const float yo = (y + u * Dd) * sil;
            yev[((size_t)b * LSEQ + (t >> 1)) * 512 + d] = (bf16_t)yo;
        }
    }
}

// ===================== host orchestration ============================
static inline void* bump(char*& p, size_t bytes)
{
    void* r = (void*)p;
    p += (bytes + 255) & ~(size_t)255;
    return r;
}

extern "C" void kernel_launch(void* const* d_in, const int* in_sizes, int n_in,
                              void* d_out, int out_size, void* d_ws, size_t ws_size,
                              hipStream_t stream)
{
    const float* x        = (const float*)d_in[0];
    const float* skip     = (const float*)d_in[1];
    const float* ln_x_w   = (const float*)d_in[2];
    const float* ln_x_b   = (const float*)d_in[3];
    const float* ln_s_w   = (const float*)d_in[4];
    const float* ln_s_b   = (const float*)d_in[5];
    const float* in_proj  = (const float*)d_in[6];   // [1024,256]
    const float* conv_w   = (const float*)d_in[7];   // [512,4]
    const float* conv_b   = (const float*)d_in[8];
    const float* x_proj   = (const float*)d_in[9];   // [48,512]
    const float* dt_projw = (const float*)d_in[10];  // [512,16]
    const float* dt_projb = (const float*)d_in[11];
    const float* A_log    = (const float*)d_in[12];  // [512,16]
    const float* Dv       = (const float*)d_in[13];
    const float* mo_w     = (const float*)d_in[14];  // [256,512]
    const float* out_w    = (const float*)d_in[15];  // [256,256]
    const float* out_b    = (const float*)d_in[16];
    float* out = (float*)d_out;

    char* ws = (char*)d_ws;
    float*  inter_f = (float*) bump(ws, (size_t)BT * CDIM * 4);
    bf16_t* inter_b = (bf16_t*)bump(ws, (size_t)BT * CDIM * 2);
    bf16_t* w_in_b  = (bf16_t*)bump(ws, (size_t)1024 * 256 * 2);
    bf16_t* w_xp_b  = (bf16_t*)bump(ws, (size_t)48 * 512 * 2);
    bf16_t* w_dt_b  = (bf16_t*)bump(ws, (size_t)512 * 32 * 2);
    bf16_t* w_mo_b  = (bf16_t*)bump(ws, (size_t)256 * 512 * 2);
    bf16_t* w_ow_b  = (bf16_t*)bump(ws, (size_t)256 * 256 * 2);
    float*  xz      = (float*) bump(ws, (size_t)BT * 1024 * 4);
    float*  u_s     = (float*) bump(ws, (size_t)BT * DIN * 4);
    bf16_t* u_b     = (bf16_t*)bump(ws, (size_t)BT * DIN * 2);
    float*  xdbl    = (float*) bump(ws, (size_t)BT * 48 * 4);
    bf16_t* dtb     = (bf16_t*)bump(ws, (size_t)BT * 32 * 2);
    float*  dts     = (float*) bump(ws, (size_t)BT * DIN * 4);
    float*  chA     = (float*) bump(ws, (size_t)BATCH * NCH * DIN * NST * 4);
    float*  chB     = (float*) bump(ws, (size_t)BATCH * NCH * DIN * NST * 4);
    float*  hst     = (float*) bump(ws, (size_t)BATCH * NCH * DIN * NST * 4);
    bf16_t* yev     = (bf16_t*)bump(ws, (size_t)BL * DIN * 2);
    bf16_t* mo_b    = (bf16_t*)bump(ws, (size_t)BL * 256 * 2);
    (void)ws_size; (void)n_in; (void)in_sizes; (void)out_size;

    // weight conversion (fp32 -> bf16, dt_proj_w zero-padded to K=32)
    f32_to_bf16<<<(1024 * 256 + 255) / 256, 256, 0, stream>>>(in_proj, w_in_b, 1024 * 256);
    f32_to_bf16<<<(48 * 512 + 255) / 256, 256, 0, stream>>>(x_proj, w_xp_b, 48 * 512);
    f32_to_bf16<<<(256 * 512 + 255) / 256, 256, 0, stream>>>(mo_w, w_mo_b, 256 * 512);
    f32_to_bf16<<<(256 * 256 + 255) / 256, 256, 0, stream>>>(out_w, w_ow_b, 256 * 256);
    pad_dtw<<<(512 * 32) / 256, 256, 0, stream>>>(dt_projw, w_dt_b);

    // 1) layernorm x/skip -> interleaved sequence (fp32 + bf16)
    ln_interleave<<<BL / 8, 256, 0, stream>>>(x, skip, ln_x_w, ln_x_b,
                                              ln_s_w, ln_s_b, inter_f, inter_b);
    // 2) in_proj GEMM: [BT,256]x[256,1024], 32x64 wave tiles
    gemm_inproj<<<(BT / 32) * (1024 / 64) / 4, 128, 0, stream>>>(inter_b, w_in_b, xz);
    // 3) depthwise causal conv + silu
    conv_silu<<<(BT * DIN) / 256, 256, 0, stream>>>(xz, conv_w, conv_b, u_s, u_b);
    // 4) x_proj GEMM: [BT,512]x[512,48], 32x48 wave tiles
    gemm_xproj<<<(BT / 32) / 4, 128, 0, stream>>>(u_b, w_xp_b, xdbl);
    // 5) dt_proj GEMM (K padded to 32) + bias + softplus
    dtprep<<<(BT * 32) / 256, 256, 0, stream>>>(xdbl, dtb);
    gemm_dtproj<<<(BT / 32) * (512 / 64) / 4, 128, 0, stream>>>(dtb, w_dt_b, dt_projb, dts);
    // 6) chunked parallel scan (A: partials, B: boundary prefix, C: finalize)
    scan_partial<<<BATCH * NCH, DIN, 0, stream>>>(dts, u_s, xdbl, A_log, chA, chB);
    scan_prefix<<<(BATCH * DIN * NST) / 256, 256, 0, stream>>>(chA, chB, hst);
    scan_final<<<BATCH * NCH, DIN, 0, stream>>>(dts, u_s, xdbl, A_log, Dv, xz, hst, yev);
    // 7) mamba_out GEMM on even rows only: [BL,512]x[512,256] -> bf16
    gemm_mo<<<(BL / 32) * (256 / 64) / 4, 128, 0, stream>>>(yev, w_mo_b, mo_b);
    // 8) out GEMM + bias + LN(x) residual: [BL,256]x[256,256] -> fp32 out
    gemm_out<<<(BL / 32) * (256 / 64) / 4, 128, 0, stream>>>(mo_b, w_ow_b, out_b, inter_f, out);
}
